// SoftMoE11_3719441678657
// MI455X (gfx1250) — compile-verified, matrix operand order checked by reading, not run on hardware
//
#include <hip/hip_runtime.h>
#include <math.h>

// ---------------- problem constants ----------------
constexpr int B  = 8;
constexpr int N  = 2048;
constexpr int D  = 1024;
constexpr int E  = 8;
constexpr int S  = 256;
constexpr int ES = E * S;          // 2048
constexpr int H  = 6 * D / E;      // 768
constexpr float EPS_NORM = 1e-12f;
constexpr float EPS_SIM  = 1e-9f;

// ---------------- WMMA types ----------------
typedef __attribute__((ext_vector_type(16))) __bf16 v16bf;
typedef __attribute__((ext_vector_type(8)))  float  v8f;

struct FragBF { uint4 lo; uint4 hi; };   // 32 bytes == 16 bf16

// Load a 16x32 bf16 A-fragment (or B^T fragment, identical layout) from a
// row-major [rows x ld] matrix (LDS or global pointer).
// ISA 7.12.2: lanes 0-15 row M=lane hold K={0..7, 16..23};
// lanes 16-31 row M=lane-16 hold K={8..15, 24..31}.
__device__ __forceinline__ v16bf load_frag(const __bf16* __restrict__ base,
                                           int ld, int row0, int k0, int lane) {
    int r  = row0 + (lane & 15);
    int kb = k0 + ((lane >> 4) << 3);          // 0 or 8
    const uint4* p = (const uint4*)(base + (size_t)r * ld + kb);
    FragBF f;
    f.lo = p[0];        // K offsets +0..+7
    f.hi = p[2];        // K offsets +16..+23 (==+32 bytes)
    return __builtin_bit_cast(v16bf, f);
}

// ---- CDNA5 async global->LDS copy (ASYNCcnt path, ISA ch.10 / 08_async_tensor) ----
__device__ __forceinline__ void async_ld16(unsigned lds_addr, unsigned long long gaddr) {
    asm volatile("global_load_async_to_lds_b128 %0, %1, off"
                 :: "v"(lds_addr), "v"(gaddr)
                 : "memory");
}
__device__ __forceinline__ void wait_async_all() {
    asm volatile("s_wait_asynccnt 0x0" ::: "memory");
}

// Block-cooperative GEMM: C_tile[64x128] per 256-thread block (8 waves, 2x4),
// A row-major [M,K], Bt row-major [Ncols,K]; K multiple of 64.
// Double-buffered LDS staging (K-slab = 64) via async loads; each wave owns a
// 32x32 C tile (8 WMMAs per slab per wave).
__device__ __forceinline__ void gemm_block(const __bf16* __restrict__ A, int lda,
                                           const __bf16* __restrict__ Bt, int ldb,
                                           int K, int mBase, int nBase,
                                           v8f c[2][2]) {
    __shared__ __bf16 shA[2][64 * 64];     // 16 KB
    __shared__ __bf16 shB[2][128 * 64];    // 32 KB
    const int t    = threadIdx.x;
    const int lane = t & 31;
    const int wid  = t >> 5;               // 8 waves (wave32)
    const int wmL  = (wid >> 2) * 32;      // local M origin (0/32)
    const int wnL  = (wid & 3) * 32;       // local N origin (0..96)
    const unsigned baseA = (unsigned)(uintptr_t)(void*)&shA[0][0];
    const unsigned baseB = (unsigned)(uintptr_t)(void*)&shB[0][0];
    const int srow = t >> 3;               // 0..31
    const int scol = (t & 7) * 8;          // 16B chunk within a 64-elem row

    // fixed per-thread LDS chunk offsets (buffer 0)
    const unsigned lA0 = baseA + (unsigned)(((srow     ) * 64 + scol) * 2);
    const unsigned lA1 = baseA + (unsigned)(((srow + 32) * 64 + scol) * 2);
    const unsigned lB0 = baseB + (unsigned)(((srow     ) * 64 + scol) * 2);
    const unsigned lB1 = baseB + (unsigned)(((srow + 32) * 64 + scol) * 2);
    const unsigned lB2 = baseB + (unsigned)(((srow + 64) * 64 + scol) * 2);
    const unsigned lB3 = baseB + (unsigned)(((srow + 96) * 64 + scol) * 2);

    // per-thread global staging addresses; advance 128 bytes (64 bf16) per slab
    unsigned long long gA0 = (unsigned long long)(uintptr_t)(A  + (size_t)(mBase + srow     ) * lda + scol);
    unsigned long long gA1 = (unsigned long long)(uintptr_t)(A  + (size_t)(mBase + srow + 32) * lda + scol);
    unsigned long long gB0 = (unsigned long long)(uintptr_t)(Bt + (size_t)(nBase + srow     ) * ldb + scol);
    unsigned long long gB1 = (unsigned long long)(uintptr_t)(Bt + (size_t)(nBase + srow + 32) * ldb + scol);
    unsigned long long gB2 = (unsigned long long)(uintptr_t)(Bt + (size_t)(nBase + srow + 64) * ldb + scol);
    unsigned long long gB3 = (unsigned long long)(uintptr_t)(Bt + (size_t)(nBase + srow + 96) * ldb + scol);

    // prologue: stage slab 0 into buffer 0
    async_ld16(lA0, gA0); async_ld16(lA1, gA1);
    async_ld16(lB0, gB0); async_ld16(lB1, gB1);
    async_ld16(lB2, gB2); async_ld16(lB3, gB3);
    gA0 += 128; gA1 += 128; gB0 += 128; gB1 += 128; gB2 += 128; gB3 += 128;

    const int nIter = K >> 6;
    int buf = 0;
    for (int it = 0; it < nIter; ++it) {
        wait_async_all();                  // own async ops done ...
        __syncthreads();                   // ... tile visible to all waves
        if (it + 1 < nIter) {              // stage next slab into other buffer
            unsigned dA = (unsigned)((buf ^ 1) * 8192);    // 64*64*2 bytes
            unsigned dB = (unsigned)((buf ^ 1) * 16384);   // 128*64*2 bytes
            async_ld16(lA0 + dA, gA0); async_ld16(lA1 + dA, gA1);
            async_ld16(lB0 + dB, gB0); async_ld16(lB1 + dB, gB1);
            async_ld16(lB2 + dB, gB2); async_ld16(lB3 + dB, gB3);
            gA0 += 128; gA1 += 128; gB0 += 128;
            gB1 += 128; gB2 += 128; gB3 += 128;
        }
        const __bf16* tA = &shA[buf][0];
        const __bf16* tB = &shB[buf][0];
#pragma unroll
        for (int ks = 0; ks < 64; ks += 32) {
            v16bf a0 = load_frag(tA, 64, wmL,      ks, lane);
            v16bf a1 = load_frag(tA, 64, wmL + 16, ks, lane);
            v16bf b0 = load_frag(tB, 64, wnL,      ks, lane);
            v16bf b1 = load_frag(tB, 64, wnL + 16, ks, lane);
            c[0][0] = __builtin_amdgcn_wmma_f32_16x16x32_bf16(false, a0, false, b0, (short)0, c[0][0], false, false);
            c[0][1] = __builtin_amdgcn_wmma_f32_16x16x32_bf16(false, a0, false, b1, (short)0, c[0][1], false, false);
            c[1][0] = __builtin_amdgcn_wmma_f32_16x16x32_bf16(false, a1, false, b0, (short)0, c[1][0], false, false);
            c[1][1] = __builtin_amdgcn_wmma_f32_16x16x32_bf16(false, a1, false, b1, (short)0, c[1][1], false, false);
        }
        buf ^= 1;
    }
}

__device__ __forceinline__ void wave_tile_origin(int& m0, int& n0, int& lane) {
    lane    = threadIdx.x & 31;
    int wid = threadIdx.x >> 5;
    m0 = blockIdx.x * 64  + (wid >> 2) * 32;
    n0 = blockIdx.y * 128 + (wid & 3) * 32;
}

// ---------------- prep kernels ----------------

// rowwise L2-normalize x over D, emit bf16 [B*N, D]
__global__ __launch_bounds__(256) void normalize_x_kernel(
        const float* __restrict__ x, __bf16* __restrict__ xn) {
    __shared__ float red[256];
    size_t row = blockIdx.x;
    int t = threadIdx.x;
    const float4 v = ((const float4*)(x + row * D))[t];
    float ss = v.x * v.x + v.y * v.y + v.z * v.z + v.w * v.w;
    red[t] = ss; __syncthreads();
    for (int s2 = 128; s2 > 0; s2 >>= 1) {
        if (t < s2) red[t] += red[t + s2];
        __syncthreads();
    }
    float inv = 1.0f / fmaxf(sqrtf(red[0]), EPS_NORM);
    __bf16* q = xn + row * D;
    q[4 * t + 0] = (__bf16)(v.x * inv);
    q[4 * t + 1] = (__bf16)(v.y * inv);
    q[4 * t + 2] = (__bf16)(v.z * inv);
    q[4 * t + 3] = (__bf16)(v.w * inv);
}

// mu [D, ES] column-normalize over D (stride ES), scale, emit muT bf16 [ES, D]
__global__ __launch_bounds__(256) void normalize_mu_kernel(
        const float* __restrict__ mu, const float* __restrict__ scale,
        __bf16* __restrict__ muT) {
    __shared__ float red[8][33];
    int esl = threadIdx.x & 31, ds = threadIdx.x >> 5;
    int es  = blockIdx.x * 32 + esl;
    float ss = 0.f;
    for (int d = ds; d < D; d += 8) {
        float v = mu[(size_t)d * ES + es];
        ss += v * v;
    }
    red[ds][esl] = ss; __syncthreads();
    if (ds == 0) {
        float tt = 0.f;
        for (int k = 0; k < 8; ++k) tt += red[k][esl];
        red[0][esl] = tt;
    }
    __syncthreads();
    float inv = scale[0] / fmaxf(sqrtf(red[0][esl]), EPS_NORM);
    for (int d = ds; d < D; d += 8)
        muT[(size_t)es * D + d] = (__bf16)(mu[(size_t)d * ES + es] * inv);
}

// generic LDS-tiled transpose: src f32 [batch][rows][cols] -> dst bf16 [batch][cols][rows]
__global__ __launch_bounds__(256) void transpose_f32_bf16_kernel(
        const float* __restrict__ src, __bf16* __restrict__ dst,
        int rows, int cols) {
    __shared__ float tile[32][33];
    int bz = blockIdx.z;
    int c0 = blockIdx.x * 32, r0 = blockIdx.y * 32;
    int tx = threadIdx.x & 31, ty = threadIdx.x >> 5;
    const float* s = src + (size_t)bz * rows * cols;
    __bf16*      d = dst + (size_t)bz * rows * cols;
    for (int k = 0; k < 4; ++k) {
        int r = r0 + ty + 8 * k;
        tile[ty + 8 * k][tx] = s[(size_t)r * cols + c0 + tx];
    }
    __syncthreads();
    for (int k = 0; k < 4; ++k) {
        int c = c0 + ty + 8 * k;
        d[(size_t)c * rows + r0 + tx] = (__bf16)tile[tx][ty + 8 * k];
    }
}

// ---------------- GEMM 1: logits = x_norm @ muT^T  (f32 out) ----------------
__global__ __launch_bounds__(256) void gemm_logits_kernel(
        const __bf16* __restrict__ xnorm, const __bf16* __restrict__ muT,
        float* __restrict__ logits) {
    int m0, n0, lane; wave_tile_origin(m0, n0, lane);
    int b = blockIdx.z;
    const __bf16* A = xnorm + (size_t)b * N * D;
    v8f c[2][2] = {{{0,0,0,0,0,0,0,0},{0,0,0,0,0,0,0,0}},
                   {{0,0,0,0,0,0,0,0},{0,0,0,0,0,0,0,0}}};
    gemm_block(A, D, muT, D, D, blockIdx.x * 64, blockIdx.y * 128, c);
    int rb = (lane >> 4) * 8, cl = lane & 15;
    float* out = logits + (size_t)b * N * ES;
    for (int i = 0; i < 2; ++i)
        for (int j = 0; j < 2; ++j)
            for (int r = 0; r < 8; ++r) {
                int row = m0 + i * 16 + rb + r;
                int col = n0 + j * 16 + cl;
                out[(size_t)row * ES + col] = c[i][j][r];
            }
}

// ---------------- softmax over slots (cw) + combine metric ----------------
__global__ __launch_bounds__(256) void softmax_slots_kernel(
        const float* __restrict__ logits, __bf16* __restrict__ cw,
        float* __restrict__ acc) {
    __shared__ float red[256];
    __shared__ float shdiag[1];
    size_t row = blockIdx.x;                 // b*N + n
    int n = (int)(row & (N - 1));
    int t = threadIdx.x;
    const float* p = logits + row * (size_t)ES;
    float v[8];
    float m = -INFINITY;
    for (int k = 0; k < 8; ++k) { v[k] = p[t + 256 * k]; m = fmaxf(m, v[k]); }
    red[t] = m; __syncthreads();
    for (int s2 = 128; s2 > 0; s2 >>= 1) { if (t < s2) red[t] = fmaxf(red[t], red[t + s2]); __syncthreads(); }
    m = red[0]; __syncthreads();
    float sum = 0.f;
    for (int k = 0; k < 8; ++k) { v[k] = expf(v[k] - m); sum += v[k]; }
    red[t] = sum; __syncthreads();
    for (int s2 = 128; s2 > 0; s2 >>= 1) { if (t < s2) red[t] += red[t + s2]; __syncthreads(); }
    float inv = 1.0f / red[0]; __syncthreads();
    float ssq = 0.f;
    __bf16* q = cw + row * ES;
    for (int k = 0; k < 8; ++k) {
        float pk = v[k] * inv;
        q[t + 256 * k] = (__bf16)pk;
        ssq += pk * pk;
        if (t + 256 * k == n) shdiag[0] = pk;      // exactly one thread hits
    }
    red[t] = ssq; __syncthreads();
    for (int s2 = 128; s2 > 0; s2 >>= 1) { if (t < s2) red[t] += red[t + s2]; __syncthreads(); }
    if (t == 0) {
        float qq = red[0] + EPS_SIM;               // row_sum = rsqrt(qq)
        atomicAdd(&acc[0], 1.0f / qq);             // sum_total contribution
        atomicAdd(&acc[1], shdiag[0] / qq);        // sum_diag contribution
    }
}

// ------------- softmax over tokens (dispatch^T) + dispatch metric -------------
__global__ __launch_bounds__(256) void softmax_tokens_kernel(
        const float* __restrict__ logits, __bf16* __restrict__ dispT,
        float* __restrict__ acc) {
    __shared__ float red[8][33];
    int esl = threadIdx.x & 31, ns = threadIdx.x >> 5;
    int b  = blockIdx.y;
    int es = blockIdx.x * 32 + esl;
    const float* base = logits + (size_t)b * N * ES + es;
    float m = -INFINITY;
    for (int nn = ns; nn < N; nn += 8) m = fmaxf(m, base[(size_t)nn * ES]);
    red[ns][esl] = m; __syncthreads();
    if (ns == 0) { float tt = red[0][esl]; for (int k = 1; k < 8; ++k) tt = fmaxf(tt, red[k][esl]); red[0][esl] = tt; }
    __syncthreads();
    m = red[0][esl]; __syncthreads();
    float sum = 0.f;
    for (int nn = ns; nn < N; nn += 8) sum += expf(base[(size_t)nn * ES] - m);
    red[ns][esl] = sum; __syncthreads();
    if (ns == 0) { float tt = 0.f; for (int k = 0; k < 8; ++k) tt += red[k][esl]; red[0][esl] = tt; }
    __syncthreads();
    float inv = 1.0f / red[0][esl]; __syncthreads();
    float ssq = 0.f;
    __bf16* out = dispT + ((size_t)b * ES + es) * N;
    for (int nn = ns; nn < N; nn += 8) {
        float pv = expf(base[(size_t)nn * ES] - m) * inv;
        out[nn] = (__bf16)pv;
        ssq += pv * pv;
    }
    red[ns][esl] = ssq; __syncthreads();
    if (ns == 0) {
        float tt = 0.f; for (int k = 0; k < 8; ++k) tt += red[k][esl];
        float qq = tt + EPS_SIM;
        float contrib = 1.0f / qq;                 // row_sum^2
        atomicAdd(&acc[2], contrib);
        if (es % (S + 1) == 0) atomicAdd(&acc[3], contrib);   // es = i*257, i<8
    }
}

// ---------------- GEMM 2: slot_in = dispT @ xT^T  (bf16 out) ----------------
__global__ __launch_bounds__(256) void gemm_slot_in_kernel(
        const __bf16* __restrict__ dispT, const __bf16* __restrict__ xT,
        __bf16* __restrict__ slot_in) {
    int m0, n0, lane; wave_tile_origin(m0, n0, lane);
    int b = blockIdx.z;
    const __bf16* A  = dispT + (size_t)b * ES * N;
    const __bf16* Bt = xT    + (size_t)b * D  * N;
    v8f c[2][2] = {{{0,0,0,0,0,0,0,0},{0,0,0,0,0,0,0,0}},
                   {{0,0,0,0,0,0,0,0},{0,0,0,0,0,0,0,0}}};
    gemm_block(A, N, Bt, N, N, blockIdx.x * 64, blockIdx.y * 128, c);
    int rb = (lane >> 4) * 8, cl = lane & 15;
    __bf16* out = slot_in + (size_t)b * ES * D;
    for (int i = 0; i < 2; ++i)
        for (int j = 0; j < 2; ++j)
            for (int r = 0; r < 8; ++r) {
                int row = m0 + i * 16 + rb + r;
                int col = n0 + j * 16 + cl;
                out[(size_t)row * D + col] = (__bf16)c[i][j][r];
            }
}

// ---------------- GEMM 3: h = gelu(slot_in @ w1 + b1)  (bf16 out) ----------------
__global__ __launch_bounds__(256) void gemm_h_kernel(
        const __bf16* __restrict__ slot_in, const __bf16* __restrict__ w1t,
        const float* __restrict__ b1, __bf16* __restrict__ hbuf) {
    int m0, n0, lane; wave_tile_origin(m0, n0, lane);
    int z = blockIdx.z, b = z >> 3, e = z & 7;
    const __bf16* A  = slot_in + ((size_t)b * ES + (size_t)e * S) * D;
    const __bf16* Bt = w1t + (size_t)e * H * D;
    v8f c[2][2] = {{{0,0,0,0,0,0,0,0},{0,0,0,0,0,0,0,0}},
                   {{0,0,0,0,0,0,0,0},{0,0,0,0,0,0,0,0}}};
    gemm_block(A, D, Bt, D, D, blockIdx.x * 64, blockIdx.y * 128, c);
    int rb = (lane >> 4) * 8, cl = lane & 15;
    __bf16* out = hbuf + (size_t)z * S * H;
    for (int i = 0; i < 2; ++i)
        for (int j = 0; j < 2; ++j)
            for (int r = 0; r < 8; ++r) {
                int row = m0 + i * 16 + rb + r;
                int col = n0 + j * 16 + cl;
                float zval = c[i][j][r] + b1[e * H + col];
                float g = 0.5f * zval * (1.0f + erff(zval * 0.70710678118654752f));
                out[(size_t)row * H + col] = (__bf16)g;
            }
}

// -------- GEMM 4: slot_outT[d, es] = (h @ w2 + b2)^T  (bf16, transposed store) --------
__global__ __launch_bounds__(256) void gemm_slot_out_kernel(
        const __bf16* __restrict__ hbuf, const __bf16* __restrict__ w2t,
        const float* __restrict__ b2, __bf16* __restrict__ soutT) {
    int m0, n0, lane; wave_tile_origin(m0, n0, lane);
    int z = blockIdx.z, b = z >> 3, e = z & 7;
    const __bf16* A  = hbuf + (size_t)z * S * H;
    const __bf16* Bt = w2t + (size_t)e * D * H;
    v8f c[2][2] = {{{0,0,0,0,0,0,0,0},{0,0,0,0,0,0,0,0}},
                   {{0,0,0,0,0,0,0,0},{0,0,0,0,0,0,0,0}}};
    gemm_block(A, H, Bt, H, H, blockIdx.x * 64, blockIdx.y * 128, c);
    int rb = (lane >> 4) * 8, cl = lane & 15;
    __bf16* out = soutT + (size_t)b * D * ES;
    for (int i = 0; i < 2; ++i)
        for (int j = 0; j < 2; ++j)
            for (int r = 0; r < 8; ++r) {
                int row = m0 + i * 16 + rb + r;      // s index
                int col = n0 + j * 16 + cl;          // d index
                float val = c[i][j][r] + b2[e * D + col];
                out[(size_t)col * ES + (size_t)e * S + row] = (__bf16)val;
            }
}

// ---------------- GEMM 5: out = cw @ soutT^T  (f32 out) ----------------
__global__ __launch_bounds__(256) void gemm_out_kernel(
        const __bf16* __restrict__ cw, const __bf16* __restrict__ soutT,
        float* __restrict__ outp) {
    int m0, n0, lane; wave_tile_origin(m0, n0, lane);
    int b = blockIdx.z;
    const __bf16* A  = cw    + (size_t)b * N * ES;
    const __bf16* Bt = soutT + (size_t)b * D * ES;
    v8f c[2][2] = {{{0,0,0,0,0,0,0,0},{0,0,0,0,0,0,0,0}},
                   {{0,0,0,0,0,0,0,0},{0,0,0,0,0,0,0,0}}};
    gemm_block(A, ES, Bt, ES, ES, blockIdx.x * 64, blockIdx.y * 128, c);
    int rb = (lane >> 4) * 8, cl = lane & 15;
    float* out = outp + (size_t)b * N * D;
    for (int i = 0; i < 2; ++i)
        for (int j = 0; j < 2; ++j)
            for (int r = 0; r < 8; ++r) {
                int row = m0 + i * 16 + rb + r;
                int col = n0 + j * 16 + cl;
                out[(size_t)row * D + col] = c[i][j][r];
            }
}

// ---------------- small kernels ----------------
__global__ void zero_acc_kernel(float* acc) {
    if (threadIdx.x < 4) acc[threadIdx.x] = 0.0f;
}

__global__ void finalize_metrics_kernel(const float* __restrict__ acc,
                                        float* __restrict__ out) {
    out[(size_t)B * N * D]     = (acc[0] - acc[1]) / (8.0f * 2048.0f * 2047.0f);
    out[(size_t)B * N * D + 1] = (acc[2] - acc[3]) / (8.0f * 8.0f * 256.0f * 2047.0f);
}

// ---------------- host launcher ----------------
extern "C" void kernel_launch(void* const* d_in, const int* in_sizes, int n_in,
                              void* d_out, int out_size, void* d_ws, size_t ws_size,
                              hipStream_t stream) {
    const float* x     = (const float*)d_in[0];
    // d_in[1] = attn_weight (unused in 'original' eval mode)
    const float* mu    = (const float*)d_in[2];
    const float* scale = (const float*)d_in[3];
    const float* w1    = (const float*)d_in[4];
    const float* b1    = (const float*)d_in[5];
    const float* w2    = (const float*)d_in[6];
    const float* b2    = (const float*)d_in[7];
    float* out = (float*)d_out;

    char* w = (char*)d_ws;
    size_t off = 0;
    float* acc = (float*)(w + off);                 off += 256;
    __bf16* xnorm = (__bf16*)(w + off);             off += (size_t)B * N * D * 2;
    __bf16* muT   = (__bf16*)(w + off);             off += (size_t)ES * D * 2;
    __bf16* xT    = (__bf16*)(w + off);             off += (size_t)B * D * N * 2;
    __bf16* dispT = (__bf16*)(w + off);             off += (size_t)B * ES * N * 2;
    __bf16* cw    = (__bf16*)(w + off);             off += (size_t)B * N * ES * 2;
    __bf16* w1t   = (__bf16*)(w + off);             off += (size_t)E * H * D * 2;
    __bf16* w2t   = (__bf16*)(w + off);             off += (size_t)E * D * H * 2;
    float* logits = (float*)(w + off);              // 134 MB region, reused below
    __bf16* slot_in = (__bf16*)(w + off);                                   // after softmaxes
    __bf16* hbuf    = (__bf16*)(w + off + (size_t)B * ES * D * 2);
    __bf16* soutT   = (__bf16*)(w + off + (size_t)B * ES * D * 2 + (size_t)B * ES * H * 2);

    zero_acc_kernel<<<1, 32, 0, stream>>>(acc);
    normalize_x_kernel<<<B * N, 256, 0, stream>>>(x, xnorm);
    transpose_f32_bf16_kernel<<<dim3(D / 32, N / 32, B), 256, 0, stream>>>(x, xT, N, D);
    transpose_f32_bf16_kernel<<<dim3(H / 32, D / 32, E), 256, 0, stream>>>(w1, w1t, D, H);
    transpose_f32_bf16_kernel<<<dim3(D / 32, H / 32, E), 256, 0, stream>>>(w2, w2t, H, D);
    normalize_mu_kernel<<<ES / 32, 256, 0, stream>>>(mu, scale, muT);

    gemm_logits_kernel<<<dim3(N / 64, ES / 128, B), 256, 0, stream>>>(xnorm, muT, logits);
    softmax_slots_kernel<<<B * N, 256, 0, stream>>>(logits, cw, acc);
    softmax_tokens_kernel<<<dim3(ES / 32, B), 256, 0, stream>>>(logits, dispT, acc);

    gemm_slot_in_kernel<<<dim3(ES / 64, D / 128, B), 256, 0, stream>>>(dispT, xT, slot_in);
    gemm_h_kernel<<<dim3(S / 64, H / 128, B * E), 256, 0, stream>>>(slot_in, w1t, b1, hbuf);
    gemm_slot_out_kernel<<<dim3(S / 64, D / 128, B * E), 256, 0, stream>>>(hbuf, w2t, b2, soutT);
    gemm_out_kernel<<<dim3(N / 64, D / 128, B), 256, 0, stream>>>(cw, soutT, out);

    finalize_metrics_kernel<<<1, 1, 0, stream>>>(acc, out);
}
